// _UnimolPELayer_91053306675331
// MI455X (gfx1250) — matrix operand order repaired and checked by reading, not compiled, not run on hardware
//
#include <hip/hip_runtime.h>
#include <hip/hip_bf16.h>

// Problem constants (from reference)
#define LQ     512
#define BATCH  16
#define DIM    512
#define NH     8
#define NP     32
#define DHEAD  64
#define FFDIM  2048
#define BHN    (BATCH * NH)   // 128
#define MROWS  (LQ * BATCH)   // 8192

typedef __attribute__((ext_vector_type(16))) __bf16 v16bf;
typedef __attribute__((ext_vector_type(8)))  float  v8f;

static_assert(sizeof(v16bf) == 32, "v16bf must be 32 bytes");

// ---------------------------------------------------------------------------
// fp32 -> bf16 round-to-nearest-even (bit pattern in ushort; buffers are ushort)
// ---------------------------------------------------------------------------
__device__ __forceinline__ unsigned short f2bf(float f) {
  unsigned int u = __float_as_uint(f);
  u += 0x7FFFu + ((u >> 16) & 1u);
  return (unsigned short)(u >> 16);
}

union FragCvt { v16bf v; uint4 q[2]; };

// A fragment: 16x32 bf16, lane m = m0 + lane%16, K per lane = {8h..8h+7} U {16+8h..16+8h+7}
__device__ __forceinline__ v16bf load_a_frag(const unsigned short* A, int lda,
                                             int m0, int k0, int lane) {
  const unsigned short* p = A + (size_t)(m0 + (lane & 15)) * lda + k0 + 8 * (lane >> 4);
  FragCvt u;
  u.q[0] = *(const uint4*)(p);
  u.q[1] = *(const uint4*)(p + 16);
  return u.v;
}

// B fragment: 32x16 bf16 from BT[n][k] (K contiguous), lane n = n0 + lane%16,
// K per lane = {8h..8h+7} U {16+8h..16+8h+7}
__device__ __forceinline__ v16bf load_b_frag(const unsigned short* BT, int ldb,
                                             int n0, int k0, int lane) {
  const unsigned short* p = BT + (size_t)(n0 + (lane & 15)) * ldb + k0 + 8 * (lane >> 4);
  FragCvt u;
  u.q[0] = *(const uint4*)(p);
  u.q[1] = *(const uint4*)(p + 16);
  return u.v;
}

// One wave computes a (16*MT) x (16*NT) fp32 tile of A[M,K] x BT[N,K]^T.
// B fragment is reused across the MT row-tiles (raises FLOPs/byte by MT on B traffic).
template <int MT, int NT>
__device__ __forceinline__ void gemm_core(const unsigned short* __restrict__ A, int lda,
                                          const unsigned short* __restrict__ BT, int ldb,
                                          int m0, int n0, int K, int lane,
                                          v8f acc[MT][NT]) {
#pragma unroll
  for (int s = 0; s < MT; ++s)
#pragma unroll
    for (int t = 0; t < NT; ++t)
#pragma unroll
      for (int i = 0; i < 8; ++i) acc[s][t][i] = 0.0f;
  for (int k0 = 0; k0 < K; k0 += 32) {
    v16bf a[MT];
#pragma unroll
    for (int s = 0; s < MT; ++s) a[s] = load_a_frag(A, lda, m0 + 16 * s, k0, lane);
#pragma unroll
    for (int t = 0; t < NT; ++t) {
      v16bf b = load_b_frag(BT, ldb, n0 + 16 * t, k0, lane);
#pragma unroll
      for (int s = 0; s < MT; ++s)
        acc[s][t] = __builtin_amdgcn_wmma_f32_16x16x32_bf16(false, a[s], false, b,
                                                            (short)0, acc[s][t], false, false);
    }
  }
}

// C/D mapping: element (r, lane) -> row m0 + 16*s + r + 8*(lane>>4), col n0 + 16*t + (lane&15)

// ---------------------------------------------------------------------------
// Weight transpose fp32 [K,N] -> bf16 [N,K]
// ---------------------------------------------------------------------------
__global__ void wt_bf16_kernel(const float* __restrict__ W, unsigned short* __restrict__ WT,
                               int K, int N) {
  size_t idx = (size_t)blockIdx.x * blockDim.x + threadIdx.x;
  if (idx >= (size_t)K * N) return;
  int n = (int)(idx / K);
  int k = (int)(idx % K);
  WT[idx] = f2bf(W[(size_t)k * N + n]);
}

// ---------------------------------------------------------------------------
// LayerNorm over last dim (D=512) -> bf16 rows
// ---------------------------------------------------------------------------
__global__ __launch_bounds__(256) void ln_bf16_kernel(const float* __restrict__ x,
                                                      const float* __restrict__ g,
                                                      const float* __restrict__ bb,
                                                      unsigned short* __restrict__ out) {
  int row = blockIdx.x;
  const float* xr = x + (size_t)row * DIM;
  float s = 0.f, s2 = 0.f;
  for (int i = threadIdx.x; i < DIM; i += 256) { float v = xr[i]; s += v; s2 += v * v; }
#pragma unroll
  for (int o = 16; o > 0; o >>= 1) { s += __shfl_down(s, o, 32); s2 += __shfl_down(s2, o, 32); }
  __shared__ float sm[16];
  int w = threadIdx.x >> 5;
  if ((threadIdx.x & 31) == 0) { sm[w] = s; sm[8 + w] = s2; }
  __syncthreads();
  if (threadIdx.x == 0) {
    float a = 0.f, c = 0.f;
    for (int i = 0; i < 8; ++i) { a += sm[i]; c += sm[8 + i]; }
    sm[0] = a * (1.0f / DIM);
    sm[8] = c * (1.0f / DIM);
  }
  __syncthreads();
  float mean = sm[0];
  float inv = rsqrtf(sm[8] - mean * mean + 1e-5f);
  unsigned short* orow = out + (size_t)row * DIM;
  for (int i = threadIdx.x; i < DIM; i += 256)
    orow[i] = f2bf((xr[i] - mean) * inv * g[i] + bb[i]);
}

// ---------------------------------------------------------------------------
// PE projection (tiny): pe_emb[P,D] @ in_proj_w + b -> pe_q/pe_k [H][P][DH] bf16
// ---------------------------------------------------------------------------
__global__ void pe_proj_kernel(const float* __restrict__ pe, const float* __restrict__ W,
                               const float* __restrict__ bias,
                               unsigned short* __restrict__ peq, unsigned short* __restrict__ pek) {
  int idx = blockIdx.x * blockDim.x + threadIdx.x;
  if (idx >= NP * 2 * DIM) return;
  int p = idx / (2 * DIM);
  int c = idx % (2 * DIM);              // [0,512) -> q col, [512,1024) -> k col
  float s = bias[c];
  const float* per = pe + (size_t)p * DIM;
  for (int d = 0; d < DIM; ++d) s += per[d] * W[(size_t)d * (3 * DIM) + c];
  int cc = c & (DIM - 1);
  int h = cc / DHEAD, dh = cc % DHEAD;
  unsigned short v = f2bf(s);
  if (c < DIM) peq[((size_t)h * NP + p) * DHEAD + dh] = v;
  else         pek[((size_t)h * NP + p) * DHEAD + dh] = v;
}

// ---------------------------------------------------------------------------
// QKV GEMM: h[8192,512] x wT_in[1536,512] -> scatter q/k [bh][L][DH], vT [bh][DH][L]
// 32x64 tiles: 256 m-tiles x 24 n-blocks = 6144 waves
// ---------------------------------------------------------------------------
__global__ __launch_bounds__(256) void qkv_gemm_kernel(const unsigned short* __restrict__ A,
                                                       const unsigned short* __restrict__ BT,
                                                       const float* __restrict__ bias,
                                                       unsigned short* __restrict__ qb,
                                                       unsigned short* __restrict__ kb,
                                                       unsigned short* __restrict__ vtb) {
  int gtid = blockIdx.x * 256 + threadIdx.x;
  int wave = gtid >> 5, lane = gtid & 31;
  const int NB = (3 * DIM) / 64;  // 24
  int mt = wave / NB, nb = wave % NB;
  int m0 = mt * 32, n0 = nb * 64;
  v8f acc[2][4];
  gemm_core<2, 4>(A, DIM, BT, DIM, m0, n0, DIM, lane, acc);
  int which = n0 / DIM;  // uniform per wave: 0=q 1=k 2=v
#pragma unroll
  for (int t = 0; t < 4; ++t) {
    int n = n0 + t * 16 + (lane & 15);
    float bs = bias[n];
    int nn = n % DIM;
    int h = nn / DHEAD, dh = nn % DHEAD;
#pragma unroll
    for (int s = 0; s < 2; ++s)
#pragma unroll
      for (int r = 0; r < 8; ++r) {
        int m = m0 + 16 * s + r + 8 * (lane >> 4);
        int l = m / BATCH, b_ = m % BATCH;
        unsigned short bv = f2bf(acc[s][t][r] + bs);
        size_t bh = (size_t)(b_ * NH + h);
        if (which == 0)      qb[(bh * LQ + l) * DHEAD + dh] = bv;
        else if (which == 1) kb[(bh * LQ + l) * DHEAD + dh] = bv;
        else                 vtb[(bh * DHEAD + dh) * LQ + l] = bv;
      }
  }
}

// ---------------------------------------------------------------------------
// sp_q[bh][P][L] = pe_q[h] (32x64) @ k[bh]^T  : one 32x64 tile per wave, 128*8 waves
// ---------------------------------------------------------------------------
__global__ __launch_bounds__(256) void spq_kernel(const unsigned short* __restrict__ peq,
                                                  const unsigned short* __restrict__ kb,
                                                  float* __restrict__ spq) {
  int gtid = blockIdx.x * 256 + threadIdx.x;
  int wave = gtid >> 5, lane = gtid & 31;
  int bh = wave >> 3, nb = wave & 7;
  v8f acc[2][4];
  gemm_core<2, 4>(peq + (size_t)(bh % NH) * NP * DHEAD, DHEAD,
                  kb + (size_t)bh * LQ * DHEAD, DHEAD, 0, nb * 64, DHEAD, lane, acc);
  float* Ch = spq + (size_t)bh * NP * LQ;
#pragma unroll
  for (int t = 0; t < 4; ++t) {
    int n = nb * 64 + t * 16 + (lane & 15);
#pragma unroll
    for (int s = 0; s < 2; ++s)
#pragma unroll
      for (int r = 0; r < 8; ++r) {
        int m = 16 * s + r + 8 * (lane >> 4);
        Ch[(size_t)m * LQ + n] = acc[s][t][r];
      }
  }
}

// sp_k[bh][L][P] = q[bh] @ pe_k[h]^T : 32x32 tiles, 128*16 = 2048 waves
__global__ __launch_bounds__(256) void spk_kernel(const unsigned short* __restrict__ qb,
                                                  const unsigned short* __restrict__ pek,
                                                  float* __restrict__ spk) {
  int gtid = blockIdx.x * 256 + threadIdx.x;
  int wave = gtid >> 5, lane = gtid & 31;
  int bh = wave >> 4, it = wave & 15;
  v8f acc[2][2];
  gemm_core<2, 2>(qb + (size_t)bh * LQ * DHEAD, DHEAD,
                  pek + (size_t)(bh % NH) * NP * DHEAD, DHEAD, it * 32, 0, DHEAD, lane, acc);
  float* Ch = spk + (size_t)bh * LQ * NP;
#pragma unroll
  for (int t = 0; t < 2; ++t) {
    int n = t * 16 + (lane & 15);
#pragma unroll
    for (int s = 0; s < 2; ++s)
#pragma unroll
      for (int r = 0; r < 8; ++r) {
        int m = it * 32 + 16 * s + r + 8 * (lane >> 4);
        Ch[(size_t)m * NP + n] = acc[s][t][r];
      }
  }
}

// ---------------------------------------------------------------------------
// Scores: t = (qk + gather(sp_q) + gather(sp_k)) / 8 ; s = t + edge -> d_out edge region
// 32x64 tiles: 128 * 16 * 8 = 16384 waves
// ---------------------------------------------------------------------------
__global__ __launch_bounds__(256) void score_kernel(const unsigned short* __restrict__ qb,
                                                    const unsigned short* __restrict__ kb,
                                                    const float* __restrict__ spq,
                                                    const float* __restrict__ spk,
                                                    const int* __restrict__ bdist,
                                                    const float* __restrict__ edge,
                                                    float* __restrict__ sout) {
  int gtid = blockIdx.x * 256 + threadIdx.x;
  int wave = gtid >> 5, lane = gtid & 31;
  int bh = wave >> 7, rem = wave & 127;
  int it = rem >> 3, jb = rem & 7;
  int b_ = bh / NH;
  v8f acc[2][4];
  gemm_core<2, 4>(qb + (size_t)bh * LQ * DHEAD, DHEAD,
                  kb + (size_t)bh * LQ * DHEAD, DHEAD, it * 32, jb * 64, DHEAD, lane, acc);
  const float* spqh = spq + (size_t)bh * NP * LQ;
  const float* spkh = spk + (size_t)bh * LQ * NP;
  const float* eh   = edge + (size_t)bh * LQ * LQ;
  float* sh         = sout + (size_t)bh * LQ * LQ;
#pragma unroll
  for (int t = 0; t < 4; ++t) {
    int j = jb * 64 + t * 16 + (lane & 15);
#pragma unroll
    for (int s = 0; s < 2; ++s)
#pragma unroll
      for (int r = 0; r < 8; ++r) {
        int i = it * 32 + 16 * s + r + 8 * (lane >> 4);
        int p = bdist[((size_t)i * LQ + j) * BATCH + b_];
        float tt = (acc[s][t][r] + spqh[(size_t)p * LQ + j] + spkh[(size_t)i * NP + p]) * 0.125f;
        sh[(size_t)i * LQ + j] = tt + eh[(size_t)i * LQ + j];
      }
  }
}

// ---------------------------------------------------------------------------
// Row softmax over 512 cols -> bf16 probs
// ---------------------------------------------------------------------------
__global__ __launch_bounds__(256) void softmax_kernel(const float* __restrict__ s,
                                                      unsigned short* __restrict__ attn) {
  size_t row = blockIdx.x;
  const float* sr = s + row * LQ;
  float mx = -3.4e38f;
  for (int i = threadIdx.x; i < LQ; i += 256) mx = fmaxf(mx, sr[i]);
#pragma unroll
  for (int o = 16; o > 0; o >>= 1) mx = fmaxf(mx, __shfl_down(mx, o, 32));
  __shared__ float sm[16];
  int w = threadIdx.x >> 5;
  if ((threadIdx.x & 31) == 0) sm[w] = mx;
  __syncthreads();
  if (threadIdx.x == 0) {
    float m2 = sm[0];
    for (int i = 1; i < 8; ++i) m2 = fmaxf(m2, sm[i]);
    sm[0] = m2;
  }
  __syncthreads();
  mx = sm[0];
  float sum = 0.f;
  for (int i = threadIdx.x; i < LQ; i += 256) sum += expf(sr[i] - mx);
#pragma unroll
  for (int o = 16; o > 0; o >>= 1) sum += __shfl_down(sum, o, 32);
  if ((threadIdx.x & 31) == 0) sm[8 + w] = sum;
  __syncthreads();
  if (threadIdx.x == 0) {
    float a = 0.f;
    for (int i = 0; i < 8; ++i) a += sm[8 + i];
    sm[8] = a;
  }
  __syncthreads();
  float inv = 1.f / sm[8];
  unsigned short* ar = attn + row * LQ;
  for (int i = threadIdx.x; i < LQ; i += 256) ar[i] = f2bf(expf(sr[i] - mx) * inv);
}

// ---------------------------------------------------------------------------
// o = attn @ v : per (bh) attn[512,512] x vT[64,512] -> o bf16 [L,B,D] layout
// 32x64 tiles: 128 * 16 = 2048 waves
// ---------------------------------------------------------------------------
__global__ __launch_bounds__(256) void av_kernel(const unsigned short* __restrict__ attn,
                                                 const unsigned short* __restrict__ vtb,
                                                 unsigned short* __restrict__ ob) {
  int gtid = blockIdx.x * 256 + threadIdx.x;
  int wave = gtid >> 5, lane = gtid & 31;
  int bh = wave >> 4, it = wave & 15;
  v8f acc[2][4];
  gemm_core<2, 4>(attn + (size_t)bh * LQ * LQ, LQ,
                  vtb + (size_t)bh * DHEAD * LQ, LQ, it * 32, 0, LQ, lane, acc);
  int b_ = bh / NH, h = bh % NH;
#pragma unroll
  for (int t = 0; t < 4; ++t) {
    int dh = t * 16 + (lane & 15);
#pragma unroll
    for (int s = 0; s < 2; ++s)
#pragma unroll
      for (int r = 0; r < 8; ++r) {
        int i = it * 32 + 16 * s + r + 8 * (lane >> 4);
        ob[((size_t)i * BATCH + b_) * DIM + h * DHEAD + dh] = f2bf(acc[s][t][r]);
      }
  }
}

// ---------------------------------------------------------------------------
// out_proj + residual: x2 = x + o @ Wout + bias  (fp32, into d_out x region)
// 256 * 8 = 2048 waves
// ---------------------------------------------------------------------------
__global__ __launch_bounds__(256) void outproj_kernel(const unsigned short* __restrict__ ob,
                                                      const unsigned short* __restrict__ WT,
                                                      const float* __restrict__ bias,
                                                      const float* __restrict__ xres,
                                                      float* __restrict__ x2) {
  int gtid = blockIdx.x * 256 + threadIdx.x;
  int wave = gtid >> 5, lane = gtid & 31;
  int mt = wave >> 3, nb = wave & 7;
  int m0 = mt * 32;
  v8f acc[2][4];
  gemm_core<2, 4>(ob, DIM, WT, DIM, m0, nb * 64, DIM, lane, acc);
#pragma unroll
  for (int t = 0; t < 4; ++t) {
    int n = nb * 64 + t * 16 + (lane & 15);
    float bs = bias[n];
#pragma unroll
    for (int s = 0; s < 2; ++s)
#pragma unroll
      for (int r = 0; r < 8; ++r) {
        int m = m0 + 16 * s + r + 8 * (lane >> 4);
        size_t idx = (size_t)m * DIM + n;
        x2[idx] = xres[idx] + acc[s][t][r] + bs;
      }
  }
}

// ---------------------------------------------------------------------------
// lin1 + exact GELU -> bf16 [8192, 2048] : 256 * 32 = 8192 waves
// ---------------------------------------------------------------------------
__global__ __launch_bounds__(256) void lin1_kernel(const unsigned short* __restrict__ h2,
                                                   const unsigned short* __restrict__ WT,
                                                   const float* __restrict__ bias,
                                                   unsigned short* __restrict__ g1) {
  int gtid = blockIdx.x * 256 + threadIdx.x;
  int wave = gtid >> 5, lane = gtid & 31;
  int mt = wave >> 5, nb = wave & 31;
  int m0 = mt * 32;
  v8f acc[2][4];
  gemm_core<2, 4>(h2, DIM, WT, DIM, m0, nb * 64, DIM, lane, acc);
#pragma unroll
  for (int t = 0; t < 4; ++t) {
    int n = nb * 64 + t * 16 + (lane & 15);
    float bs = bias[n];
#pragma unroll
    for (int s = 0; s < 2; ++s)
#pragma unroll
      for (int r = 0; r < 8; ++r) {
        int m = m0 + 16 * s + r + 8 * (lane >> 4);
        float v = acc[s][t][r] + bs;
        float ge = 0.5f * v * (1.0f + erff(v * 0.70710678118654752f));
        g1[(size_t)m * FFDIM + n] = f2bf(ge);
      }
  }
}

// ---------------------------------------------------------------------------
// lin2 + residual into d_out x region: x_out = x2 + g1 @ Wl2 + bias : 2048 waves
// ---------------------------------------------------------------------------
__global__ __launch_bounds__(256) void lin2_kernel(const unsigned short* __restrict__ g1,
                                                   const unsigned short* __restrict__ WT,
                                                   const float* __restrict__ bias,
                                                   float* __restrict__ xo) {
  int gtid = blockIdx.x * 256 + threadIdx.x;
  int wave = gtid >> 5, lane = gtid & 31;
  int mt = wave >> 3, nb = wave & 7;
  int m0 = mt * 32;
  v8f acc[2][4];
  gemm_core<2, 4>(g1, FFDIM, WT, FFDIM, m0, nb * 64, FFDIM, lane, acc);
#pragma unroll
  for (int t = 0; t < 4; ++t) {
    int n = nb * 64 + t * 16 + (lane & 15);
    float bs = bias[n];
#pragma unroll
    for (int s = 0; s < 2; ++s)
#pragma unroll
      for (int r = 0; r < 8; ++r) {
        int m = m0 + 16 * s + r + 8 * (lane >> 4);
        size_t idx = (size_t)m * DIM + n;
        xo[idx] = xo[idx] + acc[s][t][r] + bs;
      }
  }
}

// ---------------------------------------------------------------------------
extern "C" void kernel_launch(void* const* d_in, const int* in_sizes, int n_in,
                              void* d_out, int out_size, void* d_ws, size_t ws_size,
                              hipStream_t stream) {
  (void)in_sizes; (void)n_in; (void)out_size; (void)ws_size;
  const float* x      = (const float*)d_in[0];
  const float* edge   = (const float*)d_in[1];
  const int*   bdist  = (const int*)d_in[2];
  const float* w_in   = (const float*)d_in[3];
  const float* b_in   = (const float*)d_in[4];
  const float* w_out  = (const float*)d_in[5];
  const float* b_out  = (const float*)d_in[6];
  const float* pe_emb = (const float*)d_in[7];
  const float* w_l1   = (const float*)d_in[8];
  const float* b_l1   = (const float*)d_in[9];
  const float* w_l2   = (const float*)d_in[10];
  const float* b_l2   = (const float*)d_in[11];
  const float* n1g    = (const float*)d_in[12];
  const float* n1b    = (const float*)d_in[13];
  const float* n2g    = (const float*)d_in[14];
  const float* n2b    = (const float*)d_in[15];

  float* x_out = (float*)d_out;
  float* s_out = x_out + (size_t)MROWS * DIM;  // edge_out region (also softmax input)

  char* ws = (char*)d_ws;
  size_t off = 0;
  auto alloc = [&](size_t bytes) -> void* {
    void* p = ws + off;
    off = (off + bytes + 255) & ~(size_t)255;
    return p;
  };
  unsigned short* wT_in  = (unsigned short*)alloc((size_t)3 * DIM * DIM * 2);
  unsigned short* wT_out = (unsigned short*)alloc((size_t)DIM * DIM * 2);
  unsigned short* wT_l1  = (unsigned short*)alloc((size_t)FFDIM * DIM * 2);
  unsigned short* wT_l2  = (unsigned short*)alloc((size_t)DIM * FFDIM * 2);
  unsigned short* h_bf   = (unsigned short*)alloc((size_t)MROWS * DIM * 2);
  unsigned short* qb     = (unsigned short*)alloc((size_t)BHN * LQ * DHEAD * 2);
  unsigned short* kb     = (unsigned short*)alloc((size_t)BHN * LQ * DHEAD * 2);
  unsigned short* vtb    = (unsigned short*)alloc((size_t)BHN * DHEAD * LQ * 2);
  unsigned short* peq    = (unsigned short*)alloc((size_t)NH * NP * DHEAD * 2);
  unsigned short* pek    = (unsigned short*)alloc((size_t)NH * NP * DHEAD * 2);
  float*          spq    = (float*)alloc((size_t)BHN * NP * LQ * 4);
  float*          spk    = (float*)alloc((size_t)BHN * LQ * NP * 4);
  unsigned short* attn   = (unsigned short*)alloc((size_t)BHN * LQ * LQ * 2);
  unsigned short* g1     = attn;  // overlay: attn dead before lin1 output written
  unsigned short* ob     = (unsigned short*)alloc((size_t)MROWS * DIM * 2);
  unsigned short* h2     = (unsigned short*)alloc((size_t)MROWS * DIM * 2);

  // 1) weight transposes (fp32 -> bf16 [N][K])
  wt_bf16_kernel<<<((size_t)DIM * 3 * DIM + 255) / 256, 256, 0, stream>>>(w_in, wT_in, DIM, 3 * DIM);
  wt_bf16_kernel<<<((size_t)DIM * DIM + 255) / 256, 256, 0, stream>>>(w_out, wT_out, DIM, DIM);
  wt_bf16_kernel<<<((size_t)DIM * FFDIM + 255) / 256, 256, 0, stream>>>(w_l1, wT_l1, DIM, FFDIM);
  wt_bf16_kernel<<<((size_t)FFDIM * DIM + 255) / 256, 256, 0, stream>>>(w_l2, wT_l2, FFDIM, DIM);

  // 2) LN1 -> h bf16
  ln_bf16_kernel<<<MROWS, 256, 0, stream>>>(x, n1g, n1b, h_bf);

  // 3) PE projection (tiny)
  pe_proj_kernel<<<(NP * 2 * DIM + 255) / 256, 256, 0, stream>>>(pe_emb, w_in, b_in, peq, pek);

  // 4) QKV GEMM: 256 m-tiles x 24 n-blocks = 6144 waves
  qkv_gemm_kernel<<<6144 / 8, 256, 0, stream>>>(h_bf, wT_in, b_in, qb, kb, vtb);

  // 5) sp_q (128*8 = 1024 waves), sp_k (128*16 = 2048 waves)
  spq_kernel<<<1024 / 8, 256, 0, stream>>>(peq, kb, spq);
  spk_kernel<<<2048 / 8, 256, 0, stream>>>(qb, pek, spk);

  // 6) scores + bias gather + edge -> d_out edge region (128*16*8 = 16384 waves)
  score_kernel<<<16384 / 8, 256, 0, stream>>>(qb, kb, spq, spk, bdist, edge, s_out);

  // 7) softmax -> bf16 probs
  softmax_kernel<<<BHN * LQ, 256, 0, stream>>>(s_out, attn);

  // 8) attn @ v (128*16 = 2048 waves)
  av_kernel<<<2048 / 8, 256, 0, stream>>>(attn, vtb, ob);

  // 9) out_proj + residual -> x2 in d_out x region (256*8 = 2048 waves)
  outproj_kernel<<<2048 / 8, 256, 0, stream>>>(ob, wT_out, b_out, x, x_out);

  // 10) LN2 -> h2 bf16
  ln_bf16_kernel<<<MROWS, 256, 0, stream>>>(x_out, n2g, n2b, h2);

  // 11) lin1 + GELU (256*32 = 8192 waves)
  lin1_kernel<<<8192 / 8, 256, 0, stream>>>(h2, wT_l1, b_l1, g1);

  // 12) lin2 + residual into d_out x region (256*8 = 2048 waves)
  lin2_kernel<<<2048 / 8, 256, 0, stream>>>(g1, wT_l2, b_l2, x_out);
}
